// GroupQueryAttention_33234456936604
// MI455X (gfx1250) — matrix operand order repaired and checked by reading, not compiled, not run on hardware
//
#include <hip/hip_runtime.h>
#include <hip/hip_bf16.h>

#define SQ 2048            // sequence length
#define EMB 2048           // embed dim
#define HN 32              // query heads
#define KVN 8              // kv heads
#define DH 64              // head dim
#define NEG_INF (-3.402823466e38f)

typedef __attribute__((ext_vector_type(8)))  __bf16 v8bf;
typedef __attribute__((ext_vector_type(16))) __bf16 v16bf;
typedef __attribute__((ext_vector_type(8)))  float  v8f;

// ---------------- WMMA helpers (CDNA5 bf16 16x16x32, f32 accum) ----------------

static __device__ __forceinline__ v8f wmma_bf16(v16bf a, v16bf b, v8f c) {
  // 8 args: (neg_a, A, neg_b, B, c_mod, C, reuse_a, reuse_b)
  return __builtin_amdgcn_wmma_f32_16x16x32_bf16(false, a, false, b, (short)0, c,
                                                 false, false);
}

// A fragment 16x32 (MxK): lane holds row (lane&15), K-chunks at
// (lane>=16)*8 and 16+(lane>=16)*8, each 8 contiguous bf16 (16B).
static __device__ __forceinline__ v16bf ld_a(const __bf16* base, int stride,
                                             int k0, int lane) {
  const int row = lane & 15;
  const int h8  = (lane >> 4) << 3;
  const __bf16* p = base + (size_t)row * stride + k0;
  v8bf lo = *(const v8bf*)(p + h8);
  v8bf hi = *(const v8bf*)(p + 16 + h8);
  v16bf a;
#pragma unroll
  for (int i = 0; i < 8; ++i) { a[i] = lo[i]; a[i + 8] = hi[i]; }
  return a;
}

// B fragment 32x16 (KxN) with B[k,n] = Wrow[n][k]: lane holds col (lane&15),
// K-chunks at (lane>=16)*8 and 16+(lane>=16)*8, each 8 contiguous bf16 (16B).
static __device__ __forceinline__ v16bf ld_b(const __bf16* base, int stride,
                                             int k0, int lane) {
  const int n  = lane & 15;
  const int h8 = (lane >> 4) << 3;
  const __bf16* p = base + (size_t)n * stride + k0;
  v8bf lo = *(const v8bf*)(p + h8);
  v8bf hi = *(const v8bf*)(p + 16 + h8);
  v16bf b;
#pragma unroll
  for (int i = 0; i < 8; ++i) { b[i] = lo[i]; b[i + 8] = hi[i]; }
  return b;
}

// A fragment sourced from f32 memory (attention weights), converted in-register.
static __device__ __forceinline__ v16bf ld_a_f32(const float* base, int stride,
                                                 int k0, int lane) {
  const int row = lane & 15;
  const int h8  = (lane >> 4) << 3;
  const float* p = base + (size_t)row * stride + k0;
  float4 a0 = *(const float4*)(p + h8);
  float4 a1 = *(const float4*)(p + h8 + 4);
  float4 b0 = *(const float4*)(p + 16 + h8);
  float4 b1 = *(const float4*)(p + 16 + h8 + 4);
  v16bf a;
  a[0] = (__bf16)a0.x;  a[1] = (__bf16)a0.y;  a[2] = (__bf16)a0.z;  a[3] = (__bf16)a0.w;
  a[4] = (__bf16)a1.x;  a[5] = (__bf16)a1.y;  a[6] = (__bf16)a1.z;  a[7] = (__bf16)a1.w;
  a[8] = (__bf16)b0.x;  a[9] = (__bf16)b0.y;  a[10] = (__bf16)b0.z; a[11] = (__bf16)b0.w;
  a[12] = (__bf16)b1.x; a[13] = (__bf16)b1.y; a[14] = (__bf16)b1.z; a[15] = (__bf16)b1.w;
  return a;
}

// ---------------- f32 -> bf16 conversion ----------------

__global__ void cvt_f32_bf16(const float* __restrict__ src,
                             __bf16* __restrict__ dst, int n) {
  int i = blockIdx.x * blockDim.x + threadIdx.x;
  int stride = gridDim.x * blockDim.x;
  for (; i < n; i += stride) dst[i] = (__bf16)src[i];
}

// ---------------- Generic C = A @ W^T + bias (one wave = 16x64 tile) ----------------
// mode: exactly one of outb / outf / outvt is non-null.
//   outb  : bf16 output, row stride Nout
//   outf  : f32 output, row stride Nout
//   outvt : bf16 transposed per-head store Vt[kv][d][s]  (Nout ignored)

__global__ void __launch_bounds__(32)
gemm_xwT(const __bf16* __restrict__ A, const __bf16* __restrict__ W,
         const float* __restrict__ bias, int K, int Nout,
         __bf16* __restrict__ outb, float* __restrict__ outf,
         __bf16* __restrict__ outvt) {
  const int lane = threadIdx.x;
  const int m0 = blockIdx.x << 4;
  const int n0 = blockIdx.y << 6;
  const __bf16* Ab = A + (size_t)m0 * K;

  v8f c[4];
#pragma unroll
  for (int nt = 0; nt < 4; ++nt) {
    float bv = bias[n0 + (nt << 4) + (lane & 15)];
#pragma unroll
    for (int r = 0; r < 8; ++r) c[nt][r] = bv;
  }

  for (int k0 = 0; k0 < K; k0 += 32) {
    v16bf a = ld_a(Ab, K, k0, lane);
#pragma unroll
    for (int nt = 0; nt < 4; ++nt) {
      v16bf b = ld_b(W + (size_t)(n0 + (nt << 4)) * K, K, k0, lane);
      c[nt] = wmma_bf16(a, b, c[nt]);
    }
  }

  const int rbase = m0 + ((lane >> 4) << 3);
  const int nl = lane & 15;
  if (outb) {
#pragma unroll
    for (int nt = 0; nt < 4; ++nt) {
      int col = n0 + (nt << 4) + nl;
#pragma unroll
      for (int r = 0; r < 8; ++r)
        outb[(size_t)(rbase + r) * Nout + col] = (__bf16)c[nt][r];
    }
  } else if (outf) {
#pragma unroll
    for (int nt = 0; nt < 4; ++nt) {
      int col = n0 + (nt << 4) + nl;
#pragma unroll
      for (int r = 0; r < 8; ++r)
        outf[(size_t)(rbase + r) * Nout + col] = c[nt][r];
    }
  } else {
    // Vt[kv][d][s]: kv = col>>6, d = col&63; Vt stride per kv = 64*SQ = 1<<17
#pragma unroll
    for (int nt = 0; nt < 4; ++nt) {
      int col = n0 + (nt << 4) + nl;
      __bf16* vp = outvt + ((size_t)(col >> 6) << 17) + (size_t)(col & 63) * SQ;
#pragma unroll
      for (int r = 0; r < 8; ++r) vp[rbase + r] = (__bf16)c[nt][r];
    }
  }
}

// ---------------- Scores: raw scaled Q.K^T -> attn buffer + online (max,sumexp) ----------------
// One wave per (16 query rows, head). C fragment row = r + (lane>=16)*8, col = lane&15,
// so row-wise reductions are shfl_xor within each 16-lane half (masks 1,2,4,8).

__global__ void __launch_bounds__(32)
attn_scores(const __bf16* __restrict__ Qp, const __bf16* __restrict__ Kp,
            float* __restrict__ attn, float* __restrict__ stats) {
  const int lane = threadIdx.x;
  const int m0 = blockIdx.x << 4;
  const int h  = blockIdx.y;
  const int kv = h >> 2;
  const float scale = 0.125f;  // 1/sqrt(64)

  const __bf16* Ab = Qp + (size_t)m0 * EMB + h * DH;
  float* arow = attn + (size_t)h * SQ * SQ + (size_t)m0 * SQ;

  float m_run[8], l_run[8];
#pragma unroll
  for (int r = 0; r < 8; ++r) { m_run[r] = NEG_INF; l_run[r] = 0.0f; }

  const int rb = (lane >> 4) << 3;
  const int nl = lane & 15;

  for (int kt = 0; kt < SQ / 16; ++kt) {
    v8f c = {0.f, 0.f, 0.f, 0.f, 0.f, 0.f, 0.f, 0.f};
    const __bf16* Bb = Kp + (size_t)(kt << 4) * (KVN * DH) + kv * DH;
#pragma unroll
    for (int k0 = 0; k0 < DH; k0 += 32) {
      v16bf a = ld_a(Ab, EMB, k0, lane);
      v16bf b = ld_b(Bb, KVN * DH, k0, lane);
      c = wmma_bf16(a, b, c);
    }
#pragma unroll
    for (int r = 0; r < 8; ++r) c[r] *= scale;

    // per-row tile max across the 16 lanes of this half
    float mr[8];
#pragma unroll
    for (int r = 0; r < 8; ++r) {
      float v = c[r];
#pragma unroll
      for (int off = 1; off < 16; off <<= 1) v = fmaxf(v, __shfl_xor(v, off, 32));
      mr[r] = v;
    }
    // online max/sumexp update
#pragma unroll
    for (int r = 0; r < 8; ++r) {
      float mn = fmaxf(m_run[r], mr[r]);
      float e = __expf(c[r] - mn);
#pragma unroll
      for (int off = 1; off < 16; off <<= 1) e += __shfl_xor(e, off, 32);
      l_run[r] = l_run[r] * __expf(m_run[r] - mn) + e;
      m_run[r] = mn;
    }
    // raw scaled scores out (normalized in a later pass)
#pragma unroll
    for (int r = 0; r < 8; ++r)
      arow[(size_t)(rb + r) * SQ + (kt << 4) + nl] = c[r];
  }

  if (nl == 0) {
#pragma unroll
    for (int r = 0; r < 8; ++r) {
      size_t row = (size_t)h * SQ + m0 + rb + r;
      stats[row * 2]     = m_run[r];
      stats[row * 2 + 1] = l_run[r];
    }
  }
}

// ---------------- In-place softmax normalize of attn (float4 vectorized) ----------------

__global__ void attn_norm(float* __restrict__ attn, const float* __restrict__ stats,
                          size_t n4) {
  size_t i = (size_t)blockIdx.x * blockDim.x + threadIdx.x;
  size_t stride = (size_t)gridDim.x * blockDim.x;
  for (; i < n4; i += stride) {
    size_t row = i >> 9;  // (i*4)/SQ, SQ=2048
    float m   = stats[row * 2];
    float inv = 1.0f / stats[row * 2 + 1];
    float4 v = ((float4*)attn)[i];
    v.x = __expf(v.x - m) * inv;
    v.y = __expf(v.y - m) * inv;
    v.z = __expf(v.z - m) * inv;
    v.w = __expf(v.w - m) * inv;
    ((float4*)attn)[i] = v;
  }
}

// ---------------- ctx = attn @ V (A from f32 attn, B from Vt[kv][d][s]) ----------------

__global__ void __launch_bounds__(32)
attn_ctx(const float* __restrict__ attn, const __bf16* __restrict__ Vt,
         __bf16* __restrict__ ctxb) {
  const int lane = threadIdx.x;
  const int m0 = blockIdx.x << 4;
  const int h  = blockIdx.y;
  const int kv = h >> 2;
  const float* Ab = attn + (size_t)h * SQ * SQ + (size_t)m0 * SQ;
  const __bf16* Bb = Vt + ((size_t)kv << 17);  // kv * 64 * SQ

  v8f c[4];
#pragma unroll
  for (int nt = 0; nt < 4; ++nt)
#pragma unroll
    for (int r = 0; r < 8; ++r) c[nt][r] = 0.0f;

  for (int k0 = 0; k0 < SQ; k0 += 32) {
    v16bf a = ld_a_f32(Ab, SQ, k0, lane);
#pragma unroll
    for (int nt = 0; nt < 4; ++nt) {
      v16bf b = ld_b(Bb + (size_t)(nt << 4) * SQ, SQ, k0, lane);
      c[nt] = wmma_bf16(a, b, c[nt]);
    }
  }

  const int rbase = m0 + ((lane >> 4) << 3);
  const int nl = lane & 15;
#pragma unroll
  for (int nt = 0; nt < 4; ++nt) {
    int col = h * DH + (nt << 4) + nl;
#pragma unroll
    for (int r = 0; r < 8; ++r)
      ctxb[(size_t)(rbase + r) * EMB + col] = (__bf16)c[nt][r];
  }
}

// ---------------- launcher ----------------

extern "C" void kernel_launch(void* const* d_in, const int* in_sizes, int n_in,
                              void* d_out, int out_size, void* d_ws, size_t ws_size,
                              hipStream_t stream) {
  const float* q  = (const float*)d_in[0];
  const float* k  = (const float*)d_in[1];
  const float* v  = (const float*)d_in[2];
  const float* Wq = (const float*)d_in[3];
  const float* bq = (const float*)d_in[4];
  const float* Wk = (const float*)d_in[5];
  const float* bk = (const float*)d_in[6];
  const float* Wv = (const float*)d_in[7];
  const float* bv = (const float*)d_in[8];
  const float* Wo = (const float*)d_in[9];
  const float* bo = (const float*)d_in[10];

  float* out  = (float*)d_out;                      // [S, E]
  float* attn = out + (size_t)SQ * EMB;             // [H, S, S]

  const size_t M4 = (size_t)SQ * EMB;               // 4M elems
  const size_t M1 = (size_t)SQ * KVN * DH;          // 1M elems
  __bf16* wsb = (__bf16*)d_ws;
  __bf16* qb   = wsb;                               // bf16 of query
  __bf16* kb   = qb + M4;
  __bf16* vb   = kb + M4;
  __bf16* wqb  = vb + M4;
  __bf16* wkb  = wqb + M4;                          // 512x2048
  __bf16* wvb  = wkb + M1;
  __bf16* wob  = wvb + M1;
  __bf16* Qp   = wob + M4;                          // projected Q [S,E]
  __bf16* Kp   = Qp + M4;                           // projected K [S,512]
  __bf16* Vt   = Kp + M1;                           // V transposed [KV][D][S]
  __bf16* ctxb = Vt + M1;                           // context [S,E]
  float*  stats = (float*)(ctxb + M4);              // [H*S][2]

  // 1) f32 -> bf16 conversions
  cvt_f32_bf16<<<1024, 256, 0, stream>>>(q,  qb,  (int)M4);
  cvt_f32_bf16<<<1024, 256, 0, stream>>>(k,  kb,  (int)M4);
  cvt_f32_bf16<<<1024, 256, 0, stream>>>(v,  vb,  (int)M4);
  cvt_f32_bf16<<<1024, 256, 0, stream>>>(Wq, wqb, (int)M4);
  cvt_f32_bf16<<<1024, 256, 0, stream>>>(Wk, wkb, (int)M1);
  cvt_f32_bf16<<<1024, 256, 0, stream>>>(Wv, wvb, (int)M1);
  cvt_f32_bf16<<<1024, 256, 0, stream>>>(Wo, wob, (int)M4);

  // 2) projections (x @ W^T + b) via bf16 WMMA
  gemm_xwT<<<dim3(SQ / 16, EMB / 64), 32, 0, stream>>>(
      qb, wqb, bq, EMB, EMB, Qp, nullptr, nullptr);
  gemm_xwT<<<dim3(SQ / 16, (KVN * DH) / 64), 32, 0, stream>>>(
      kb, wkb, bk, EMB, KVN * DH, Kp, nullptr, nullptr);
  gemm_xwT<<<dim3(SQ / 16, (KVN * DH) / 64), 32, 0, stream>>>(
      vb, wvb, bv, EMB, 0, nullptr, nullptr, Vt);

  // 3) scores + online softmax stats (raw scaled scores -> attn buffer)
  attn_scores<<<dim3(SQ / 16, HN), 32, 0, stream>>>(Qp, Kp, attn, stats);

  // 4) normalize attn in place (this is the reference attn_weights output)
  attn_norm<<<8192, 256, 0, stream>>>(attn, stats, (size_t)HN * SQ * SQ / 4);

  // 5) ctx = attn @ V
  attn_ctx<<<dim3(SQ / 16, HN), 32, 0, stream>>>(attn, Vt, ctxb);

  // 6) output = ctx @ Wo^T + bo (f32 epilogue into d_out)
  gemm_xwT<<<dim3(SQ / 16, EMB / 64), 32, 0, stream>>>(
      ctxb, wob, bo, EMB, EMB, nullptr, out, nullptr);
}
